// Attention_16269336117792
// MI455X (gfx1250) — matrix operand order repaired and checked by reading, not compiled, not run on hardware
//
#include <hip/hip_runtime.h>
#include <hip/hip_bf16.h>

// ---------------------------------------------------------------------------
// Attention (B=8, N=2048, C=1024), f32 in/out, bf16 WMMA internally.
//   qkv = x @ w_qkv^T            (16384 x 3072 x K=1024)
//   S   = (q*scale) @ k^T        (per batch: 2048 x 2048 x K=1024)
//   P   = softmax_rows(S)        (bf16)
//   O   = P @ v                  (per batch: 2048 x 1024 x K=2048), stored O^T
//   out = reshape(O^T) @ w_proj^T + b_proj   (16384 x 1024 x K=1024)
// Matmuls: v_wmma_f32_16x16x32_bf16; A/B tiles double-buffered in LDS via
// GLOBAL_LOAD_ASYNC_TO_LDS_B128 (ASYNCcnt) so tile movement overlaps WMMA.
// K-loop is unrolled 2x so the buffer ping-pong is structural (no dynamic
// buffer select -> no accumulator shuffling).
// ---------------------------------------------------------------------------

#define DIMC 1024
#define SEQ  2048
#define BATCH 8
#define MTOT (BATCH * SEQ)   // 16384

#define LDS_STRIDE 40        // 32 K-elements + 8 pad (80B rows: conflict-free)

typedef __attribute__((ext_vector_type(16))) __bf16 v16bf;
typedef __attribute__((ext_vector_type(8)))  float  v8f;

union FragAB { v16bf v; uint4 u[2]; };   // 32 bytes: 16 bf16 per lane

__device__ __forceinline__ unsigned short f2bf(float f) {
  unsigned int x = __float_as_uint(f);
  x = (x + 0x7FFFu + ((x >> 16) & 1u)) >> 16;   // round-to-nearest-even
  return (unsigned short)x;
}

__device__ __forceinline__ v8f vzero8() {
  v8f z = {0.f, 0.f, 0.f, 0.f, 0.f, 0.f, 0.f, 0.f};
  return z;
}

// Async copy of one 16B chunk global -> LDS (no VGPR return, ASYNCcnt).
__device__ __forceinline__ void async_cp16(unsigned lds_byte_addr, const void* gptr) {
  asm volatile("global_load_async_to_lds_b128 %0, %1, off"
               :: "v"(lds_byte_addr), "v"(gptr)
               : "memory");
}
__device__ __forceinline__ void wait_async_le4() {
  asm volatile("s_wait_asynccnt 0x4" ::: "memory");
}
__device__ __forceinline__ void wait_async_le0() {
  asm volatile("s_wait_asynccnt 0x0" ::: "memory");
}

__device__ __forceinline__ unsigned lds_addr_of(const void* p) {
  // LDS aperture is 4GB-aligned: low 32 bits of a flat shared pointer are the
  // wave-relative LDS byte offset expected by *_async_to_lds VDST.
  return (unsigned)(unsigned long long)p;
}

// ---------------------------------------------------------------------------
// Block-level GEMM core: 256 threads (8 waves as 4x2), block tile 128(M)x128(N),
// K-step 32, K assumed a multiple of 64.  A row-major (K contig), B-source
// N-major (K contig, i.e. W[n][k]).  Per wave: 32x64 tile -> acc[2][4].
// ---------------------------------------------------------------------------
__device__ __forceinline__ void block_mma_128x128(
    const unsigned short* __restrict__ A, int lda,
    const unsigned short* __restrict__ B, int ldb,
    int blockRow, int blockCol, int K, v8f acc[2][4]) {
  __shared__ unsigned short Asm[2][128 * LDS_STRIDE];
  __shared__ unsigned short Bsm[2][128 * LDS_STRIDE];

  const int tid  = threadIdx.x;
  const int lane = tid & 31;
  const int w    = tid >> 5;
  const int wm   = w & 3;        // 4 wave-rows  -> 32 M each
  const int wn   = w >> 2;       // 2 wave-cols  -> 64 N each
  const int h    = lane >> 4;
  const int l16  = lane & 15;

#pragma unroll
  for (int tm = 0; tm < 2; ++tm)
#pragma unroll
    for (int tn = 0; tn < 4; ++tn) acc[tm][tn] = vzero8();

  // Each tile = 128 rows x 32 elems = 512 x 16B chunks; thread owns chunks
  // {tid, tid+256}; chunk c: row = c>>2, 16B sub-chunk = c&3.
  const int c0row = tid >> 2,         c0sub = tid & 3;
  const int c1row = (tid + 256) >> 2, c1sub = (tid + 256) & 3;

  auto issue_step = [&](int buf, int k0) {
    async_cp16(lds_addr_of(&Asm[buf][c0row * LDS_STRIDE + c0sub * 8]),
               A + (size_t)(blockRow + c0row) * lda + k0 + c0sub * 8);
    async_cp16(lds_addr_of(&Bsm[buf][c0row * LDS_STRIDE + c0sub * 8]),
               B + (size_t)(blockCol + c0row) * ldb + k0 + c0sub * 8);
    async_cp16(lds_addr_of(&Asm[buf][c1row * LDS_STRIDE + c1sub * 8]),
               A + (size_t)(blockRow + c1row) * lda + k0 + c1sub * 8);
    async_cp16(lds_addr_of(&Bsm[buf][c1row * LDS_STRIDE + c1sub * 8]),
               B + (size_t)(blockCol + c1row) * ldb + k0 + c1sub * 8);
  };

  // Fragment ds_load + 8x WMMA from one resident buffer.
  auto compute_step = [&](const unsigned short* As, const unsigned short* Bs) {
    FragAB a[2], b[4];
#pragma unroll
    for (int tm = 0; tm < 2; ++tm) {
      // A frag 16x32: lane row = l16, K = 8*h + e (e<8), then +16.
      const unsigned short* p = As + (wm * 32 + tm * 16 + l16) * LDS_STRIDE + 8 * h;
      a[tm].u[0] = *(const uint4*)(p);
      a[tm].u[1] = *(const uint4*)(p + 16);
    }
#pragma unroll
    for (int tn = 0; tn < 4; ++tn) {
      // B frag 32x16: lane col = l16, K = 8*h + e (e<8), then +16.
      const unsigned short* p = Bs + (wn * 64 + tn * 16 + l16) * LDS_STRIDE + 8 * h;
      b[tn].u[0] = *(const uint4*)(p);
      b[tn].u[1] = *(const uint4*)(p + 16);
    }
#pragma unroll
    for (int tm = 0; tm < 2; ++tm)
#pragma unroll
      for (int tn = 0; tn < 4; ++tn)
        acc[tm][tn] = __builtin_amdgcn_wmma_f32_16x16x32_bf16(
            false, a[tm].v, false, b[tn].v, (short)0, acc[tm][tn], false, false);
  };

  const int nsteps = K >> 5;          // multiple of 2 for all call sites
  issue_step(0, 0);

  for (int kt = 0; kt < nsteps; kt += 2) {
    // ---- phase 0: compute buffer 0, prefetch into buffer 1 ----
    issue_step(1, (kt + 1) << 5);     // kt+1 < nsteps always (nsteps even)
    wait_async_le4();                 // 4 newest outstanding = buf1; buf0 done
    __syncthreads();                  // buf0 visible to all waves
    compute_step(&Asm[0][0], &Bsm[0][0]);
    __syncthreads();                  // all reads of buf0 done before refill

    // ---- phase 1: compute buffer 1, prefetch into buffer 0 ----
    if (kt + 2 < nsteps) {
      issue_step(0, (kt + 2) << 5);
      wait_async_le4();
    } else {
      wait_async_le0();
    }
    __syncthreads();                  // buf1 visible to all waves
    compute_step(&Asm[1][0], &Bsm[1][0]);
    __syncthreads();                  // all reads of buf1 done before refill
  }
}

// --------------------------- f32 -> bf16 convert ---------------------------
__global__ void cvt_kernel(const float4* __restrict__ in, ushort4* __restrict__ out, int n4) {
  int i = blockIdx.x * blockDim.x + threadIdx.x;
  const int st = gridDim.x * blockDim.x;
  for (; i < n4; i += st) {
    float4 f = in[i];
    ushort4 o;
    o.x = f2bf(f.x); o.y = f2bf(f.y); o.z = f2bf(f.z); o.w = f2bf(f.w);
    out[i] = o;
  }
}

// ------------------------------- QKV GEMM ----------------------------------
// qkv = x @ w_qkv^T.  d<1024 -> q (scale 1/32 folded), d<2048 -> k,
// else -> v stored TRANSPOSED per batch: vt[b][c][m].
__global__ __launch_bounds__(256) void gemm_qkv_kernel(
    const unsigned short* __restrict__ xb, const unsigned short* __restrict__ wqkv,
    unsigned short* __restrict__ qb, unsigned short* __restrict__ kb,
    unsigned short* __restrict__ vtb) {
  const int lane = threadIdx.x & 31;
  const int w    = threadIdx.x >> 5;
  v8f acc[2][4];
  block_mma_128x128(xb, DIMC, wqkv, DIMC, blockIdx.x * 128, blockIdx.y * 128, DIMC, acc);
  const int row0 = blockIdx.x * 128 + (w & 3) * 32;
  const int col0 = blockIdx.y * 128 + (w >> 2) * 64;
  const int h = lane >> 4, n = lane & 15;
#pragma unroll
  for (int tm = 0; tm < 2; ++tm)
#pragma unroll
    for (int tn = 0; tn < 4; ++tn) {
      const int d = col0 + tn * 16 + n;
#pragma unroll
      for (int r = 0; r < 8; ++r) {
        const int   mi = row0 + tm * 16 + h * 8 + r;
        const float v  = acc[tm][tn][r];
        if (d < DIMC) {
          qb[(size_t)mi * DIMC + d] = f2bf(v * 0.03125f);  // 1/sqrt(1024)
        } else if (d < 2 * DIMC) {
          kb[(size_t)mi * DIMC + (d - DIMC)] = f2bf(v);
        } else {
          const int bb = mi >> 11, nn = mi & (SEQ - 1);
          vtb[((size_t)bb * DIMC + (d - 2 * DIMC)) * SEQ + nn] = f2bf(v);
        }
      }
    }
}

// ----------------------------- scores S = q k^T ----------------------------
__global__ __launch_bounds__(256) void gemm_scores_kernel(
    const unsigned short* __restrict__ qb, const unsigned short* __restrict__ kb,
    float* __restrict__ S) {
  const int b = blockIdx.z;
  const unsigned short* A = qb + (size_t)b * SEQ * DIMC;
  const unsigned short* B = kb + (size_t)b * SEQ * DIMC;
  float* C = S + (size_t)b * SEQ * SEQ;
  const int lane = threadIdx.x & 31;
  const int w    = threadIdx.x >> 5;
  v8f acc[2][4];
  block_mma_128x128(A, DIMC, B, DIMC, blockIdx.x * 128, blockIdx.y * 128, DIMC, acc);
  const int row0 = blockIdx.x * 128 + (w & 3) * 32;
  const int col0 = blockIdx.y * 128 + (w >> 2) * 64;
  const int h = lane >> 4, n = lane & 15;
#pragma unroll
  for (int tm = 0; tm < 2; ++tm)
#pragma unroll
    for (int tn = 0; tn < 4; ++tn)
#pragma unroll
      for (int r = 0; r < 8; ++r)
        C[(size_t)(row0 + tm * 16 + h * 8 + r) * SEQ + col0 + tn * 16 + n] = acc[tm][tn][r];
}

// ------------------------------ row softmax --------------------------------
__global__ __launch_bounds__(256) void softmax_kernel(const float* __restrict__ S,
                                                      unsigned short* __restrict__ P) {
  __shared__ float red[256];
  const size_t row = blockIdx.x;
  const float* s = S + row * SEQ;
  unsigned short* p = P + row * SEQ;
  float v[8];
  float mx = -3.402823466e38f;
#pragma unroll
  for (int i = 0; i < 8; ++i) { v[i] = s[threadIdx.x + 256 * i]; mx = fmaxf(mx, v[i]); }
  red[threadIdx.x] = mx; __syncthreads();
  for (int o = 128; o > 0; o >>= 1) {
    if ((int)threadIdx.x < o) red[threadIdx.x] = fmaxf(red[threadIdx.x], red[threadIdx.x + o]);
    __syncthreads();
  }
  mx = red[0]; __syncthreads();
  float sum = 0.f;
#pragma unroll
  for (int i = 0; i < 8; ++i) { v[i] = __expf(v[i] - mx); sum += v[i]; }
  red[threadIdx.x] = sum; __syncthreads();
  for (int o = 128; o > 0; o >>= 1) {
    if ((int)threadIdx.x < o) red[threadIdx.x] += red[threadIdx.x + o];
    __syncthreads();
  }
  const float inv = 1.0f / red[0];
#pragma unroll
  for (int i = 0; i < 8; ++i) p[threadIdx.x + 256 * i] = f2bf(v[i] * inv);
}

// --------------------------- O = P v (store O^T) ---------------------------
// B-source is vt[b] ([c][m], m contiguous) so B fragments are contiguous.
// Output stored transposed: ot[b][c][m] (== reference swapaxes+reshape quirk);
// the 8 rows per lane are memory-consecutive -> one 16B packed bf16 store.
__global__ __launch_bounds__(256) void gemm_o_kernel(
    const unsigned short* __restrict__ P, const unsigned short* __restrict__ vtb,
    unsigned short* __restrict__ otb) {
  const int b = blockIdx.z;
  const unsigned short* A = P + (size_t)b * SEQ * SEQ;
  const unsigned short* B = vtb + (size_t)b * DIMC * SEQ;
  unsigned short* O = otb + (size_t)b * DIMC * SEQ;   // [c][m]
  const int lane = threadIdx.x & 31;
  const int w    = threadIdx.x >> 5;
  v8f acc[2][4];
  block_mma_128x128(A, SEQ, B, SEQ, blockIdx.x * 128, blockIdx.y * 128, SEQ, acc);
  const int row0 = blockIdx.x * 128 + (w & 3) * 32;    // q rows
  const int col0 = blockIdx.y * 128 + (w >> 2) * 64;   // channels
  const int h = lane >> 4, n = lane & 15;
#pragma unroll
  for (int tm = 0; tm < 2; ++tm)
#pragma unroll
    for (int tn = 0; tn < 4; ++tn) {
      const int c  = col0 + tn * 16 + n;
      const int mb = row0 + tm * 16 + h * 8;
      unsigned int w0 = (unsigned)f2bf(acc[tm][tn][0]) | ((unsigned)f2bf(acc[tm][tn][1]) << 16);
      unsigned int w1 = (unsigned)f2bf(acc[tm][tn][2]) | ((unsigned)f2bf(acc[tm][tn][3]) << 16);
      unsigned int w2 = (unsigned)f2bf(acc[tm][tn][4]) | ((unsigned)f2bf(acc[tm][tn][5]) << 16);
      unsigned int w3 = (unsigned)f2bf(acc[tm][tn][6]) | ((unsigned)f2bf(acc[tm][tn][7]) << 16);
      uint4 pk = {w0, w1, w2, w3};
      *(uint4*)(O + (size_t)c * SEQ + mb) = pk;
    }
}

// --------------------- out = reshape(O^T) @ w_proj^T + b -------------------
__global__ __launch_bounds__(256) void gemm_proj_kernel(
    const unsigned short* __restrict__ otb, const unsigned short* __restrict__ wpb,
    const float* __restrict__ bias, float* __restrict__ out) {
  const int lane = threadIdx.x & 31;
  const int w    = threadIdx.x >> 5;
  v8f acc[2][4];
  block_mma_128x128(otb, DIMC, wpb, DIMC, blockIdx.x * 128, blockIdx.y * 128, DIMC, acc);
  const int row0 = blockIdx.x * 128 + (w & 3) * 32;
  const int col0 = blockIdx.y * 128 + (w >> 2) * 64;
  const int h = lane >> 4, n = lane & 15;
#pragma unroll
  for (int tm = 0; tm < 2; ++tm)
#pragma unroll
    for (int tn = 0; tn < 4; ++tn) {
      const int d = col0 + tn * 16 + n;
      const float bv = bias[d];
#pragma unroll
      for (int r = 0; r < 8; ++r)
        out[(size_t)(row0 + tm * 16 + h * 8 + r) * DIMC + d] = acc[tm][tn][r] + bv;
    }
}

// ------------------------------- launcher ----------------------------------
extern "C" void kernel_launch(void* const* d_in, const int* in_sizes, int n_in,
                              void* d_out, int out_size, void* d_ws, size_t ws_size,
                              hipStream_t stream) {
  const float* x      = (const float*)d_in[0];
  const float* w_qkv  = (const float*)d_in[1];
  const float* w_proj = (const float*)d_in[2];
  const float* b_proj = (const float*)d_in[3];
  float* out = (float*)d_out;
  (void)in_sizes; (void)n_in; (void)out_size; (void)ws_size;

  // Workspace carve (~360 MB total), 256B aligned chunks.
  char* base = (char*)d_ws;
  size_t off = 0;
  auto carve = [&](size_t bytes) -> char* {
    char* p = base + off;
    off += (bytes + 255) & ~(size_t)255;
    return p;
  };
  unsigned short* xb  = (unsigned short*)carve((size_t)MTOT * DIMC * 2);      // 32 MB
  unsigned short* wqb = (unsigned short*)carve((size_t)3 * DIMC * DIMC * 2);  //  6 MB
  unsigned short* wpb = (unsigned short*)carve((size_t)DIMC * DIMC * 2);      //  2 MB
  unsigned short* qb  = (unsigned short*)carve((size_t)MTOT * DIMC * 2);      // 32 MB
  unsigned short* kb  = (unsigned short*)carve((size_t)MTOT * DIMC * 2);      // 32 MB
  unsigned short* vtb = (unsigned short*)carve((size_t)MTOT * DIMC * 2);      // 32 MB
  float*          S   = (float*)carve((size_t)BATCH * SEQ * SEQ * 4);         // 128 MB
  unsigned short* P   = (unsigned short*)carve((size_t)BATCH * SEQ * SEQ * 2);//  64 MB
  unsigned short* otb = (unsigned short*)carve((size_t)MTOT * DIMC * 2);      //  32 MB

  cvt_kernel<<<2048, 256, 0, stream>>>((const float4*)x,      (ushort4*)xb,  MTOT * DIMC / 4);
  cvt_kernel<<<512,  256, 0, stream>>>((const float4*)w_qkv,  (ushort4*)wqb, 3 * DIMC * DIMC / 4);
  cvt_kernel<<<256,  256, 0, stream>>>((const float4*)w_proj, (ushort4*)wpb, DIMC * DIMC / 4);

  gemm_qkv_kernel   <<<dim3(MTOT / 128, 3 * DIMC / 128),   256, 0, stream>>>(xb, wqb, qb, kb, vtb);
  gemm_scores_kernel<<<dim3(SEQ / 128, SEQ / 128, BATCH),  256, 0, stream>>>(qb, kb, S);
  softmax_kernel    <<<BATCH * SEQ,                        256, 0, stream>>>(S, P);
  gemm_o_kernel     <<<dim3(SEQ / 128, DIMC / 128, BATCH), 256, 0, stream>>>(P, vtb, otb);
  gemm_proj_kernel  <<<dim3(MTOT / 128, DIMC / 128),       256, 0, stream>>>(otb, wpb, b_proj, out);
}
